// GGMLLayer_58849641890140
// MI455X (gfx1250) — matrix operand order repaired and checked by reading, not compiled, not run on hardware
//
#include <hip/hip_runtime.h>
#include <hip/hip_bf16.h>

typedef __attribute__((ext_vector_type(16))) _Float16 v16h;
typedef __attribute__((ext_vector_type(8)))  _Float16 v8h;
typedef __attribute__((ext_vector_type(8)))  float    v8f;
typedef __attribute__((ext_vector_type(4)))  float    v4f;

#define OUT_DIM 4096
#define IN_DIM  4096
#define NTOK    8192   // 4 * 2048 tokens
#define NBLK    128
#define RANK    16

// ---------------- Stage 1: x (f32) -> f16 ----------------
__global__ __launch_bounds__(256)
void cvt_x_kernel(const float* __restrict__ x, _Float16* __restrict__ x16) {
    size_t idx = ((size_t)blockIdx.x * blockDim.x + threadIdx.x) * 8;
    v4f a = *(const v4f*)(x + idx);
    v4f b = *(const v4f*)(x + idx + 4);
    v8h o;
#pragma unroll
    for (int t = 0; t < 4; ++t) { o[t] = (_Float16)a[t]; o[t + 4] = (_Float16)b[t]; }
    *(v8h*)(x16 + idx) = o;
}

// ---------------- Stage 2: dequant + LoRA patch -> W16 (row-major [OUT][IN]) ----------------
__global__ __launch_bounds__(256)
void dequant_lora_kernel(const int* __restrict__ q,
                         const float* __restrict__ scales,
                         const float* __restrict__ lora_up,
                         const float* __restrict__ lora_down,
                         _Float16* __restrict__ w16) {
    int idx = blockIdx.x * blockDim.x + threadIdx.x;   // one element of [OUT][IN]
    int o   = idx >> 12;                                // / IN_DIM
    int i   = idx & (IN_DIM - 1);
    int blk = i >> 5;
    float v = (float)(q[idx] - 128) * scales[o * NBLK + blk];
    float acc = 0.0f;
#pragma unroll
    for (int r = 0; r < RANK; ++r)
        acc += lora_up[o * RANK + r] * lora_down[r * IN_DIM + i];
    w16[idx] = (_Float16)(v + acc);
}

// ---------------- Stage 3: WMMA GEMM  y[M=8192][N=4096] = X16 * W16^T + bias ----------------
// Block: 128 threads = 4 waves in a 2x2 grid; each wave owns a 64x64 output tile
// -> 16 v_wmma per K-step per wave vs 16 ds_load_b128.
// Global->LDS staging: gfx1250 async copies (ASYNCcnt), 3-deep pipeline with
// partial-count waits (s_wait_asynccnt 8) so each tile copy gets ~2 compute
// phases of latency hiding.
#define BM   128
#define BN   128
#define BK   32
#define KT   (IN_DIM / BK)   // 128 k-steps
#define LDSP 40              // padded LDS row stride in halves (32 + 8)

__device__ __forceinline__ void async_ld_b128(const _Float16* lds_ptr, const _Float16* gptr) {
    unsigned int loff = (unsigned int)(unsigned long long)lds_ptr;  // low 32b = LDS offset
    unsigned long long gaddr = (unsigned long long)gptr;
    asm volatile("global_load_async_to_lds_b128 %0, %1, off"
                 :: "v"(loff), "v"(gaddr)
                 : "memory");
}

__device__ __forceinline__ void wait_async_le8() {
    asm volatile("s_wait_asynccnt 0x8" ::: "memory");
}
__device__ __forceinline__ void wait_async_0() {
    asm volatile("s_wait_asynccnt 0x0" ::: "memory");
}

__device__ __forceinline__ v16h join16(v8h lo, v8h hi) {
    v16h r;
#pragma unroll
    for (int t = 0; t < 8; ++t) { r[t] = lo[t]; r[t + 8] = hi[t]; }
    return r;
}

__global__ __launch_bounds__(128)
void gemm_wmma_kernel(const _Float16* __restrict__ X,
                      const _Float16* __restrict__ W,
                      const float* __restrict__ bias,
                      float* __restrict__ Y) {
    __shared__ _Float16 Xs[3][BM * LDSP];
    __shared__ _Float16 Ws[3][BN * LDSP];

    const int tid  = threadIdx.x;
    const int lane = tid & 31;
    const int wave = tid >> 5;
    const int m0 = blockIdx.y * BM;
    const int n0 = blockIdx.x * BN;
    const int wm = (wave & 1) * 64;   // 2 waves along M (each 64 rows)
    const int wn = (wave >> 1) * 64;  // 2 waves along N (each 64 cols)

    // loader: 512 chunks of 8 halves per tile, 4 per thread (128 threads)
    int lrow[4], lcol[4];
#pragma unroll
    for (int t = 0; t < 4; ++t) {
        const int c = tid + t * 128;
        lrow[t] = c >> 2;            // 4 chunks per 32-half row
        lcol[t] = (c & 3) * 8;
    }

    // issue one tile's async batch: 8 x b128 per thread (4 X + 4 W)
    auto issue_tile = [&](int kb, int buf) {
        const int k0 = kb * BK;
#pragma unroll
        for (int t = 0; t < 4; ++t) {
            async_ld_b128(&Xs[buf][lrow[t] * LDSP + lcol[t]],
                          X + (size_t)(m0 + lrow[t]) * IN_DIM + k0 + lcol[t]);
            async_ld_b128(&Ws[buf][lrow[t] * LDSP + lcol[t]],
                          W + (size_t)(n0 + lrow[t]) * IN_DIM + k0 + lcol[t]);
        }
    };

    v8f acc[4][4];
#pragma unroll
    for (int mi = 0; mi < 4; ++mi)
#pragma unroll
        for (int ni = 0; ni < 4; ++ni)
#pragma unroll
            for (int t = 0; t < 8; ++t) acc[mi][ni][t] = 0.0f;

    // prologue: stage tiles 0 and 1; wait until tile 0's batch retired (<=8 left)
    issue_tile(0, 0);
    issue_tile(1, 1);
    wait_async_le8();
    __syncthreads();

    const int asel = (lane < 16) ? 0 : 8;    // A: K-half selector per ISA layout
    const int bsel = (lane < 16) ? 0 : 8;    // B: K-half selector per ISA layout (same as A)
    const int frow = lane & 15;

    int cur = 0;
    for (int kt = 0; kt < KT; ++kt) {
        const bool deep = (kt + 2) < KT;
        if (deep) {
            const int nbuf = (cur >= 1) ? cur - 1 : cur + 2;   // (cur+2)%3
            issue_tile(kt + 2, nbuf);
        }

        // A fragments (4 along M)
        v16h afrag[4];
#pragma unroll
        for (int mi = 0; mi < 4; ++mi) {
            const _Float16* ap = &Xs[cur][(wm + mi * 16 + frow) * LDSP + asel];
            afrag[mi] = join16(*(const v8h*)(ap), *(const v8h*)(ap + 16));
        }
        // B fragments (4 along N)
        v16h bfrag[4];
#pragma unroll
        for (int ni = 0; ni < 4; ++ni) {
            const _Float16* bp = &Ws[cur][(wn + ni * 16 + frow) * LDSP + bsel];
            bfrag[ni] = join16(*(const v8h*)(bp), *(const v8h*)(bp + 16));
        }

        // 16 WMMAs: D = A*B + C  (f32 accumulate)
#pragma unroll
        for (int mi = 0; mi < 4; ++mi)
#pragma unroll
            for (int ni = 0; ni < 4; ++ni)
                acc[mi][ni] = __builtin_amdgcn_wmma_f32_16x16x32_f16(
                    false, afrag[mi], false, bfrag[ni],
                    (short)0, acc[mi][ni], false, false);

        // in-order ASYNCcnt: <=8 outstanding guarantees batches kt & kt+1 retired;
        // near the tail drain fully so the last tiles are resident.
        if (deep) wait_async_le8();
        else      wait_async_0();
        __syncthreads();

        cur = (cur + 1 == 3) ? 0 : cur + 1;
    }

    // Epilogue: C/D layout — VGPR r: lanes 0-15 -> (M=r, N=lane), lanes 16-31 -> (M=8+r, N=lane-16)
    const int ncol = lane & 15;
    const int rsel = (lane >> 4) * 8;
#pragma unroll
    for (int ni = 0; ni < 4; ++ni) {
        const int n = n0 + wn + ni * 16 + ncol;
        const float bv = bias[n];
#pragma unroll
        for (int mi = 0; mi < 4; ++mi) {
#pragma unroll
            for (int r = 0; r < 8; ++r) {
                const int m = m0 + wm + mi * 16 + rsel + r;
                Y[(size_t)m * OUT_DIM + n] = acc[mi][ni][r] + bv;
            }
        }
    }
}

extern "C" void kernel_launch(void* const* d_in, const int* in_sizes, int n_in,
                              void* d_out, int out_size, void* d_ws, size_t ws_size,
                              hipStream_t stream) {
    const float* x         = (const float*)d_in[0];
    const int*   q_weight  = (const int*)d_in[1];
    const float* scales    = (const float*)d_in[2];
    const float* lora_up   = (const float*)d_in[3];
    const float* lora_down = (const float*)d_in[4];
    const float* bias      = (const float*)d_in[5];
    float* y = (float*)d_out;

    _Float16* x16 = (_Float16*)d_ws;                               // 8192*4096*2 = 64 MiB
    _Float16* w16 = x16 + (size_t)NTOK * IN_DIM;                   // 4096*4096*2 = 32 MiB

    // Stage 1: x -> f16
    {
        const int total = NTOK * IN_DIM / 8;     // 8 elems/thread
        cvt_x_kernel<<<total / 256, 256, 0, stream>>>(x, x16);
    }
    // Stage 2: dequant + LoRA -> W16
    {
        const int total = OUT_DIM * IN_DIM;      // 1 elem/thread
        dequant_lora_kernel<<<total / 256, 256, 0, stream>>>(q_weight, scales, lora_up,
                                                             lora_down, w16);
    }
    // Stage 3: WMMA GEMM + bias
    {
        dim3 grid(OUT_DIM / BN, NTOK / BM);      // 32 x 64 blocks of 128 threads
        gemm_wmma_kernel<<<grid, 128, 0, stream>>>(x16, w16, bias, y);
    }
}
